// MultiHeadAttention_223338299668
// MI455X (gfx1250) — matrix operand order repaired and checked by reading, not compiled, not run on hardware
//
#include <hip/hip_runtime.h>
#include <stdint.h>

#define DIMC     1024
#define NHEADS   16
#define HDIM     64
#define BATCH    2
#define SEQ      2048
#define MROWS    (BATCH * SEQ)          // 4096 tokens
#define QK_SCALE 0.125f                 // 64^-0.5
#define NAT2EXP2 1.4426950408889634f    // log2(e)

typedef __attribute__((ext_vector_type(16))) __bf16 v16bf;
typedef __attribute__((ext_vector_type(8)))  float  v8f;
typedef __attribute__((ext_vector_type(4)))  unsigned int v4u;
typedef __attribute__((ext_vector_type(8)))  unsigned int v8u;

union FragU { v16bf v; uint4 q[2]; };

__device__ __forceinline__ unsigned short f2bf(float f) {
  unsigned int u = __float_as_uint(f);
  u += 0x7FFFu + ((u >> 16) & 1u);      // round-to-nearest-even
  return (unsigned short)(u >> 16);
}

// A fragment (16xK row-major tile). lane: row = lane%16.
// elems 0..7 -> k = kb..kb+7 ; elems 8..15 -> k = kb+16..kb+23 ; kb=(lane>>4)*8.
__device__ __forceinline__ v16bf load_frag_a(const unsigned short* p) {
  FragU f;
  f.q[0] = *(const uint4*)(p);
  f.q[1] = *(const uint4*)(p + 16);
  return f.v;
}
// B fragment (Kx16 tile). lane: col = lane%16.
// elems 0..7 -> k = kb..kb+7 ; elems 8..15 -> k = kb+16..kb+23 ; kb=(lane>>4)*8.
__device__ __forceinline__ v16bf load_frag_b(const unsigned short* p) {
  FragU f;
  const int hb = ((threadIdx.x & 31) >> 4) * 8;
  f.q[0] = *(const uint4*)(p - hb);
  f.q[1] = *(const uint4*)(p - hb + 16);
  return f.v;
}

__device__ __forceinline__ v8f wmma_bf16(v16bf a, v16bf b, v8f c) {
  return __builtin_amdgcn_wmma_f32_16x16x32_bf16(
      false, a, false, b, (short)0, c, false, false);
}

// ---- Tensor Data Mover: async 2-D bf16 tile load, global -> LDS, with HW
// padding (pad_amount = 4 DWORDs every 2^(pad_interval_code+1) DWORDs).
// D# groups built from wave-uniform values; 2-D tensors use groups 0/1 only.
__device__ __forceinline__ void tdm_load_tile_2d(const unsigned short* gsrc,
                                                 unsigned lds_byte_off,
                                                 unsigned dim0, unsigned dim1,
                                                 unsigned stride0,
                                                 unsigned pad_interval_code) {
  const unsigned long long ga = (unsigned long long)(uintptr_t)gsrc;
  const v4u g0 = {
      1u,                                               // count=1 (valid user D#)
      lds_byte_off,                                     // D#.lds_addr
      (unsigned)ga,                                     // global_addr[31:0]
      ((unsigned)(ga >> 32) & 0x01FFFFFFu) | (2u << 30) // global_addr[56:32] | type=2
  };
  const v8u g1 = {
      (1u << 16) | (1u << 20) | (pad_interval_code << 22) | (3u << 25),
      //  data_size=2B | pad_enable | pad_interval      | pad_amount=4 DW
      (dim0 & 0xFFFFu) << 16,                           // tensor_dim0[15:0]
      (dim0 >> 16) | ((dim1 & 0xFFFFu) << 16),          // td0[31:16] | tensor_dim1[15:0]
      (dim1 >> 16) | ((dim0 & 0xFFFFu) << 16),          // td1[31:16] | tile_dim0
      dim1 & 0xFFFFu,                                   // tile_dim1  (tile_dim2=0)
      stride0,                                          // tensor_dim0_stride[31:0]
      0u, 0u
  };
  asm volatile("tensor_load_to_lds %0, %1" :: "s"(g0), "s"(g1) : "memory");
}

// ---------------------------------------------------------------- convert
__global__ __launch_bounds__(256)
void cvt_f32_bf16(const float* __restrict__ in, unsigned short* __restrict__ out, int n) {
  int i = (blockIdx.x * blockDim.x + threadIdx.x) * 4;
  if (i < n) {
    float4 v = *(const float4*)(in + i);
    ushort4 o;
    o.x = f2bf(v.x); o.y = f2bf(v.y); o.z = f2bf(v.z); o.w = f2bf(v.w);
    *(ushort4*)(out + i) = o;
  }
}

// ---------------------------------------------------------------- QKV GEMM
__global__ __launch_bounds__(256)
void qkv_gemm(const unsigned short* __restrict__ Xb,
              const unsigned short* __restrict__ Wb,
              unsigned short* __restrict__ Qo,
              unsigned short* __restrict__ Ko,
              unsigned short* __restrict__ VTo) {
  const int lane = threadIdx.x & 31;
  const int w    = threadIdx.x >> 5;
  const int wm   = w & 1, wn = w >> 1;
  const int rowBase = blockIdx.y * 128 + wm * 64;   // 4 M-tiles
  const int colBase = blockIdx.x * 128 + wn * 32;   // 2 N-tiles
  const int ln  = lane & 15;
  const int ka  = (lane >> 4) * 8;
  const int kbb = (lane >> 4) * 16;
  const int hi  = lane >> 4;

  v8f acc[4][2] = {};
  for (int k0 = 0; k0 < DIMC; k0 += 32) {
    v16bf a0 = load_frag_a(Xb + (size_t)(rowBase +      ln) * DIMC + k0 + ka);
    v16bf a1 = load_frag_a(Xb + (size_t)(rowBase + 16 + ln) * DIMC + k0 + ka);
    v16bf a2 = load_frag_a(Xb + (size_t)(rowBase + 32 + ln) * DIMC + k0 + ka);
    v16bf a3 = load_frag_a(Xb + (size_t)(rowBase + 48 + ln) * DIMC + k0 + ka);
    v16bf b0 = load_frag_b(Wb + (size_t)(colBase +      ln) * DIMC + k0 + kbb);
    v16bf b1 = load_frag_b(Wb + (size_t)(colBase + 16 + ln) * DIMC + k0 + kbb);
    acc[0][0] = wmma_bf16(a0, b0, acc[0][0]);
    acc[1][0] = wmma_bf16(a1, b0, acc[1][0]);
    acc[2][0] = wmma_bf16(a2, b0, acc[2][0]);
    acc[3][0] = wmma_bf16(a3, b0, acc[3][0]);
    acc[0][1] = wmma_bf16(a0, b1, acc[0][1]);
    acc[1][1] = wmma_bf16(a1, b1, acc[1][1]);
    acc[2][1] = wmma_bf16(a2, b1, acc[2][1]);
    acc[3][1] = wmma_bf16(a3, b1, acc[3][1]);
  }
  for (int tn = 0; tn < 2; ++tn) {
    const int col   = colBase + tn * 16 + ln;   // 0..3071
    const int which = col >> 10;                // 0=q 1=k 2=v
    const int f     = col & (DIMC - 1);
    const int h     = f >> 6;
    const int d     = f & (HDIM - 1);
    for (int tm = 0; tm < 4; ++tm) {
      for (int r = 0; r < 8; ++r) {
        const int row = rowBase + tm * 16 + hi * 8 + r;
        const int b   = row >> 11;
        const int n   = row & (SEQ - 1);
        const float val0 = acc[tm][tn][r];
        const size_t bh  = (size_t)b * NHEADS + h;
        if (which == 0) {
          Qo[(bh * SEQ + n) * HDIM + d] = f2bf(val0 * QK_SCALE);
        } else if (which == 1) {
          Ko[(bh * SEQ + n) * HDIM + d] = f2bf(val0);
        } else {
          VTo[(bh * HDIM + d) * SEQ + n] = f2bf(val0);
        }
      }
    }
  }
}

// ---------------------------------------------------------------- flash attention
// Q,K: [B*H, SEQ, HDIM] bf16 (Q pre-scaled). VT: [B*H, HDIM, SEQ] bf16.
// K/V^T tiles are double-buffered in LDS and staged by the Tensor Data Mover
// (wave0 -> K tile, wave1 -> V^T tile), one DMA ahead of compute.
__global__ __launch_bounds__(128)
void flash_attn(const unsigned short* __restrict__ Q,
                const unsigned short* __restrict__ K,
                const unsigned short* __restrict__ VT,
                unsigned short* __restrict__ Oo) {
  const int tid  = threadIdx.x;
  const int lane = tid & 31;
  const int w    = tid >> 5;                   // 4 waves x 16 q-rows
  const int bh   = blockIdx.x;                 // 0..31
  const int qtile = blockIdx.y * 64 + w * 16;
  const int ln  = lane & 15;
  const int ka  = (lane >> 4) * 8;
  const int kbb = (lane >> 4) * 16;
  const int hi  = lane >> 4;

  const unsigned short* Qb = Q  + (size_t)bh * SEQ * HDIM;
  const unsigned short* Kb = K  + (size_t)bh * SEQ * HDIM;
  const unsigned short* Vb = VT + (size_t)bh * HDIM * SEQ;

  // Row strides are odd multiples of 16B (72/40 ushorts) -> conflict-free b128
  // fragment reads; produced directly by TDM pad_interval/pad_amount.
  __shared__ __attribute__((aligned(16))) unsigned short ldsK[2][32][72]; // [keys][dims]
  __shared__ __attribute__((aligned(16))) unsigned short ldsV[2][64][40]; // [dims][keys]
  __shared__ __attribute__((aligned(16))) unsigned short pbuf[4][16][40]; // P relayout

  // prologue: DMA tile 0 into buffer 0 while we load Q fragments
  if (w == 0) {
    tdm_load_tile_2d(Kb, (unsigned)(uintptr_t)&ldsK[0][0][0], HDIM, 32, HDIM, 4);
  } else if (w == 1) {
    tdm_load_tile_2d(Vb, (unsigned)(uintptr_t)&ldsV[0][0][0], 32, HDIM, SEQ, 3);
  }

  const v16bf qa0 = load_frag_a(Qb + (size_t)(qtile + ln) * HDIM + 0  + ka);
  const v16bf qa1 = load_frag_a(Qb + (size_t)(qtile + ln) * HDIM + 32 + ka);

  v8f o0 = {}, o1 = {}, o2 = {}, o3 = {};
  float m[8], l[8];
  for (int r = 0; r < 8; ++r) { m[r] = -1e30f; l[r] = 0.0f; }

  for (int j = 0; j < SEQ; j += 32) {
    const int cur = (j >> 5) & 1;
    // issue next tile's DMA, then wait for the current tile (<=1 in flight)
    if (j + 32 < SEQ) {
      if (w == 0) {
        tdm_load_tile_2d(Kb + (size_t)(j + 32) * HDIM,
                         (unsigned)(uintptr_t)&ldsK[cur ^ 1][0][0], HDIM, 32, HDIM, 4);
      } else if (w == 1) {
        tdm_load_tile_2d(Vb + (j + 32),
                         (unsigned)(uintptr_t)&ldsV[cur ^ 1][0][0], 32, HDIM, SEQ, 3);
      }
      if (w < 2) asm volatile("s_wait_tensorcnt 0x1" ::: "memory");
    } else {
      if (w < 2) asm volatile("s_wait_tensorcnt 0x0" ::: "memory");
    }
    __syncthreads();   // (A) current tile visible to all waves

    // ---- S = Q @ K^T (two 16x16 tiles covering keys j..j+31)
    v16bf k00 = load_frag_b(&ldsK[cur][     ln][0  + kbb]);
    v16bf k01 = load_frag_b(&ldsK[cur][     ln][32 + kbb]);
    v16bf k10 = load_frag_b(&ldsK[cur][16 + ln][0  + kbb]);
    v16bf k11 = load_frag_b(&ldsK[cur][16 + ln][32 + kbb]);
    v8f s0 = {}, s1 = {};
    s0 = wmma_bf16(qa0, k00, s0);  s0 = wmma_bf16(qa1, k01, s0);
    s1 = wmma_bf16(qa0, k10, s1);  s1 = wmma_bf16(qa1, k11, s1);

    // ---- online softmax (rows in 16-lane halves; xor-reduce 1,2,4,8)
    for (int r = 0; r < 8; ++r) {
      float mx = fmaxf(s0[r], s1[r]);
      mx = fmaxf(mx, __shfl_xor(mx, 1, 32));
      mx = fmaxf(mx, __shfl_xor(mx, 2, 32));
      mx = fmaxf(mx, __shfl_xor(mx, 4, 32));
      mx = fmaxf(mx, __shfl_xor(mx, 8, 32));
      const float mn = fmaxf(m[r], mx);
      const float alpha = exp2f((m[r] - mn) * NAT2EXP2);
      const float p0 = exp2f((s0[r] - mn) * NAT2EXP2);
      const float p1 = exp2f((s1[r] - mn) * NAT2EXP2);
      float rs = p0 + p1;
      rs += __shfl_xor(rs, 1, 32);
      rs += __shfl_xor(rs, 2, 32);
      rs += __shfl_xor(rs, 4, 32);
      rs += __shfl_xor(rs, 8, 32);
      l[r] = l[r] * alpha + rs;
      m[r] = mn;
      o0[r] *= alpha; o1[r] *= alpha; o2[r] *= alpha; o3[r] *= alpha;
      pbuf[w][hi * 8 + r][ln]      = f2bf(p0);
      pbuf[w][hi * 8 + r][16 + ln] = f2bf(p1);
    }
    FragU pf;
    pf.q[0] = *(const uint4*)&pbuf[w][ln][ka];
    pf.q[1] = *(const uint4*)&pbuf[w][ln][ka + 16];

    // ---- O += P @ V
    v16bf v0 = load_frag_b(&ldsV[cur][     ln][kbb]);
    v16bf v1 = load_frag_b(&ldsV[cur][16 + ln][kbb]);
    v16bf v2 = load_frag_b(&ldsV[cur][32 + ln][kbb]);
    v16bf v3 = load_frag_b(&ldsV[cur][48 + ln][kbb]);
    o0 = wmma_bf16(pf.v, v0, o0);
    o1 = wmma_bf16(pf.v, v1, o1);
    o2 = wmma_bf16(pf.v, v2, o2);
    o3 = wmma_bf16(pf.v, v3, o3);
    __syncthreads();   // (B) reads done before this buffer is re-DMA'd
  }

  const int b = bh >> 4, h = bh & (NHEADS - 1);
  for (int r = 0; r < 8; ++r) {
    const float inv = 1.0f / l[r];
    const int n = qtile + hi * 8 + r;
    const size_t base = ((size_t)b * SEQ + n) * DIMC + h * HDIM;
    Oo[base +  0 + ln] = f2bf(o0[r] * inv);
    Oo[base + 16 + ln] = f2bf(o1[r] * inv);
    Oo[base + 32 + ln] = f2bf(o2[r] * inv);
    Oo[base + 48 + ln] = f2bf(o3[r] * inv);
  }
}

// ---------------------------------------------------------------- proj GEMM (+bias)
__global__ __launch_bounds__(256)
void proj_gemm(const unsigned short* __restrict__ Ab,
               const unsigned short* __restrict__ Wb,
               const float* __restrict__ bias,
               float* __restrict__ Out) {
  const int lane = threadIdx.x & 31;
  const int w    = threadIdx.x >> 5;
  const int wm   = w & 1, wn = w >> 1;
  const int rowBase = blockIdx.y * 128 + wm * 64;
  const int colBase = blockIdx.x * 128 + wn * 32;
  const int ln  = lane & 15;
  const int ka  = (lane >> 4) * 8;
  const int kbb = (lane >> 4) * 16;
  const int hi  = lane >> 4;

  v8f acc[4][2] = {};
  for (int k0 = 0; k0 < DIMC; k0 += 32) {
    v16bf a0 = load_frag_a(Ab + (size_t)(rowBase +      ln) * DIMC + k0 + ka);
    v16bf a1 = load_frag_a(Ab + (size_t)(rowBase + 16 + ln) * DIMC + k0 + ka);
    v16bf a2 = load_frag_a(Ab + (size_t)(rowBase + 32 + ln) * DIMC + k0 + ka);
    v16bf a3 = load_frag_a(Ab + (size_t)(rowBase + 48 + ln) * DIMC + k0 + ka);
    v16bf b0 = load_frag_b(Wb + (size_t)(colBase +      ln) * DIMC + k0 + kbb);
    v16bf b1 = load_frag_b(Wb + (size_t)(colBase + 16 + ln) * DIMC + k0 + kbb);
    acc[0][0] = wmma_bf16(a0, b0, acc[0][0]);
    acc[1][0] = wmma_bf16(a1, b0, acc[1][0]);
    acc[2][0] = wmma_bf16(a2, b0, acc[2][0]);
    acc[3][0] = wmma_bf16(a3, b0, acc[3][0]);
    acc[0][1] = wmma_bf16(a0, b1, acc[0][1]);
    acc[1][1] = wmma_bf16(a1, b1, acc[1][1]);
    acc[2][1] = wmma_bf16(a2, b1, acc[2][1]);
    acc[3][1] = wmma_bf16(a3, b1, acc[3][1]);
  }
  for (int tn = 0; tn < 2; ++tn) {
    const int col = colBase + tn * 16 + ln;
    const float bv = bias[col];
    for (int tm = 0; tm < 4; ++tm) {
      for (int r = 0; r < 8; ++r) {
        const int row = rowBase + tm * 16 + hi * 8 + r;
        Out[(size_t)row * DIMC + col] = acc[tm][tn][r] + bv;
      }
    }
  }
}

// ---------------------------------------------------------------- launch
extern "C" void kernel_launch(void* const* d_in, const int* in_sizes, int n_in,
                              void* d_out, int out_size, void* d_ws, size_t ws_size,
                              hipStream_t stream) {
  const float* x      = (const float*)d_in[0];   // [2,2048,1024]
  const float* w_qkv  = (const float*)d_in[1];   // [3072,1024]
  const float* w_proj = (const float*)d_in[2];   // [1024,1024]
  const float* b_proj = (const float*)d_in[3];   // [1024]

  unsigned short* xb     = (unsigned short*)d_ws;                 // 4096*1024
  unsigned short* wqkvb  = xb     + (size_t)MROWS * DIMC;         // 3072*1024
  unsigned short* wprojb = wqkvb  + (size_t)3 * DIMC * DIMC;      // 1024*1024
  unsigned short* qs     = wprojb + (size_t)DIMC * DIMC;          // [B*H,SEQ,HDIM]
  unsigned short* ks     = qs     + (size_t)MROWS * DIMC;
  unsigned short* vts    = ks     + (size_t)MROWS * DIMC;         // [B*H,HDIM,SEQ]
  unsigned short* ao     = vts    + (size_t)MROWS * DIMC;         // [4096,1024]

  const int nx = MROWS * DIMC, nqkv = 3 * DIMC * DIMC, npj = DIMC * DIMC;
  cvt_f32_bf16<<<nx   / 1024, 256, 0, stream>>>(x,      xb,     nx);
  cvt_f32_bf16<<<nqkv / 1024, 256, 0, stream>>>(w_qkv,  wqkvb,  nqkv);
  cvt_f32_bf16<<<npj  / 1024, 256, 0, stream>>>(w_proj, wprojb, npj);

  qkv_gemm<<<dim3(24, 32), 256, 0, stream>>>(xb, wqkvb, qs, ks, vts);
  flash_attn<<<dim3(BATCH * NHEADS, SEQ / 64), 128, 0, stream>>>(qs, ks, vts, ao);
  proj_gemm<<<dim3(8, 32), 256, 0, stream>>>(ao, wprojb, b_proj, (float*)d_out);
}
